// AGCNODEFunc_54709293416501
// MI455X (gfx1250) — compile-verified, matrix operand order checked by reading, not run on hardware
//
#include <hip/hip_runtime.h>
#include <hip/hip_bf16.h>
#include <math.h>
#include <stdint.h>

#define B_ 4
#define N_ 4096
#define F_ 64
#define E_ 16

typedef __attribute__((ext_vector_type(2)))  float    v2f;
typedef __attribute__((ext_vector_type(8)))  float    v8f;
typedef __attribute__((ext_vector_type(4)))  _Float16 h4;
typedef __attribute__((ext_vector_type(8)))  _Float16 h8;
typedef __attribute__((ext_vector_type(16))) _Float16 v16h;

// ---------------------------------------------------------------------------
// CDNA5 async global->LDS copy (ASYNCcnt-tracked, no VGPR data round-trip).
// VDST carries the 32-bit LDS byte offset (generic LDS addr keeps it in [31:0]).
// ---------------------------------------------------------------------------
__device__ __forceinline__ void async_copy_b128(unsigned lds_off, const void* g)
{
    asm volatile("global_load_async_to_lds_b128 %0, %1, off"
                 :: "v"(lds_off), "v"(g) : "memory");
}
__device__ __forceinline__ void wait_async0()
{
    asm volatile("s_wait_asynccnt 0x0" ::: "memory");
}

// ===========================================================================
// Split-FP16 WMMA GEMM (dominant path):
//   C[M,Nc] = A[M,K] * B[K,Nc],  A given as Ah+Al (row-major, K-contig),
//   B given TRANSPOSED as Bh+Bl with shape [Nc,K] (K-contig).
//   C ~= Ah*Bh + Ah*Bl + Al*Bh, f32 accumulate (v_wmma_f32_16x16x32_f16).
// BM=BN=128, BK=64; 256 thr = 8 waves (4x2); wave tile 32x64 = 2x4 accums.
// Double-buffered LDS staged by global_load_async_to_lds_b128.
// ===========================================================================
#define GM 128
#define GN 128
#define GK 64
#define LDK 72        // halves: 144B row stride -> walks LDS banks, 16B aligned
#define TILE_H (GM * LDK)

__device__ __forceinline__ v16h load_frag_f16(const _Float16* row_base, int koff)
{
    // fragment = two contiguous 8-half chunks: K = koff..+7 and K = koff+16..+23
    h8 c0 = *reinterpret_cast<const h8*>(row_base + koff);
    h8 c1 = *reinterpret_cast<const h8*>(row_base + koff + 16);
    return __builtin_shufflevector(c0, c1, 0, 1, 2, 3, 4, 5, 6, 7,
                                   8, 9, 10, 11, 12, 13, 14, 15);
}

__global__ __launch_bounds__(256)
void wmma_gemm_f16x3(const _Float16* __restrict__ Ah, const _Float16* __restrict__ Al,
                     const _Float16* __restrict__ Bh, const _Float16* __restrict__ Bl,
                     float* __restrict__ C, int M, int Nc, int K)
{
    // [buffer][matrix: Ah, Al, Bh, Bl][tile]  = 2*4*9216*2B = 144 KB
    __shared__ _Float16 sb[2][4][TILE_H];

    const int tid   = threadIdx.x;
    const int lane  = tid & 31;
    const int wave  = tid >> 5;
    const int waveM = wave & 3;   // 4 wave rows of 32
    const int waveN = wave >> 2;  // 2 wave cols of 64
    const int l15   = lane & 15;
    const int half  = lane >> 4;

    const int bM = blockIdx.y * GM;
    const int bN = blockIdx.x * GN;

    const _Float16* gsrc[4] = { Ah, Al, Bh, Bl };

    auto stage = [&](int buf, int k0) {
        #pragma unroll
        for (int i = 0; i < 4; ++i) {
            const int idx = tid + i * 256;      // 0..1023 chunks per tile
            const int row = idx >> 3;           // 8 chunks of 8 halves per row
            const int cv  = (idx & 7) << 3;
            #pragma unroll
            for (int m = 0; m < 4; ++m) {
                const int base = (m < 2) ? bM : bN;
                const size_t g = (size_t)(base + row) * K + k0 + cv;
                async_copy_b128(
                    (unsigned)(uintptr_t)&sb[buf][m][row * LDK + cv],
                    &gsrc[m][g]);
            }
        }
    };

    v8f acc[2][4] = {};

    const int nstage = K / GK;
    stage(0, 0);

    for (int it = 0; it < nstage; ++it) {
        const int cur = it & 1;
        wait_async0();          // our copies into sb[cur] have landed in LDS
        __syncthreads();        // everyone's copies done; sb[cur^1] free to refill
        if (it + 1 < nstage) stage(cur ^ 1, (it + 1) * GK);

        const _Float16* Ash = sb[cur][0];
        const _Float16* Asl = sb[cur][1];
        const _Float16* Bsh = sb[cur][2];
        const _Float16* Bsl = sb[cur][3];

        #pragma unroll
        for (int kk = 0; kk < GK; kk += 32) {
            const int koff = kk + half * 8;
            v16h ah[2], al[2], bh[4], bl[4];
            #pragma unroll
            for (int mt = 0; mt < 2; ++mt) {
                const int row = (waveM * 32 + mt * 16 + l15) * LDK;
                ah[mt] = load_frag_f16(&Ash[row], koff);
                al[mt] = load_frag_f16(&Asl[row], koff);
            }
            #pragma unroll
            for (int nt = 0; nt < 4; ++nt) {
                const int row = (waveN * 64 + nt * 16 + l15) * LDK;
                bh[nt] = load_frag_f16(&Bsh[row], koff);
                bl[nt] = load_frag_f16(&Bsl[row], koff);
            }
            #pragma unroll
            for (int mt = 0; mt < 2; ++mt)
                #pragma unroll
                for (int nt = 0; nt < 4; ++nt) {
                    acc[mt][nt] = __builtin_amdgcn_wmma_f32_16x16x32_f16(
                        false, ah[mt], false, bh[nt], (short)0, acc[mt][nt], false, false);
                    acc[mt][nt] = __builtin_amdgcn_wmma_f32_16x16x32_f16(
                        false, ah[mt], false, bl[nt], (short)0, acc[mt][nt], false, false);
                    acc[mt][nt] = __builtin_amdgcn_wmma_f32_16x16x32_f16(
                        false, al[mt], false, bh[nt], (short)0, acc[mt][nt], false, false);
                }
        }
    }

    #pragma unroll
    for (int mt = 0; mt < 2; ++mt)
        #pragma unroll
        for (int nt = 0; nt < 4; ++nt) {
            const int n = bN + waveN * 64 + nt * 16 + l15;
            #pragma unroll
            for (int r = 0; r < 8; ++r) {
                const int m = bM + waveM * 32 + mt * 16 + r + half * 8;
                C[(size_t)m * Nc + n] = acc[mt][nt][r];
            }
        }
}

// ===========================================================================
// f32 WMMA GEMM (small contractions, Nc=64): C = A[M,K] * B[K,Nc], row-major.
// mode 1: C = sc[0]*acc + bi[0]
// ===========================================================================
#define BM 128
#define BN 64
#define BK 32
#define LDA_S (BK + 2)

__global__ __launch_bounds__(256)
void wmma_gemm_f32(const float* __restrict__ A, const float* __restrict__ Bm,
                   float* __restrict__ C, int M, int Nc, int K,
                   int mode, const float* __restrict__ sc, const float* __restrict__ bi)
{
    __shared__ float As[BM * LDA_S];
    __shared__ float Bs[BK * BN];

    const int tid   = threadIdx.x;
    const int lane  = tid & 31;
    const int wave  = tid >> 5;
    const int waveM = wave & 3;
    const int waveN = wave >> 2;
    const int l15   = lane & 15;
    const int half  = lane >> 4;

    const int bM = blockIdx.y * BM;
    const int bN = blockIdx.x * BN;

    v8f acc[2][2] = {};

    for (int k0 = 0; k0 < K; k0 += BK) {
        #pragma unroll
        for (int i = 0; i < 4; ++i) {
            int idx = tid + i * 256;
            int row = idx >> 3;
            int cv  = (idx & 7) << 2;
            float4 v = *reinterpret_cast<const float4*>(
                &A[(size_t)(bM + row) * K + k0 + cv]);
            float* dst = &As[row * LDA_S + cv];
            dst[0] = v.x; dst[1] = v.y; dst[2] = v.z; dst[3] = v.w;
        }
        #pragma unroll
        for (int i = 0; i < 2; ++i) {
            int idx = tid + i * 256;
            int row = idx >> 4;
            int cv  = (idx & 15) << 2;
            *reinterpret_cast<float4*>(&Bs[row * BN + cv]) =
                *reinterpret_cast<const float4*>(
                    &Bm[(size_t)(k0 + row) * Nc + bN + cv]);
        }
        if (k0 + BK < K) {
            __builtin_prefetch(&A[(size_t)(bM + (tid >> 3)) * K + k0 + BK], 0, 1);
            __builtin_prefetch(&Bm[(size_t)(k0 + BK + (tid >> 4)) * Nc + bN], 0, 1);
        }
        __syncthreads();

        #pragma unroll
        for (int kk = 0; kk < BK; kk += 4) {
            const int ka = kk + half * 2;
            v2f afr[2], bfr[2];
            #pragma unroll
            for (int mt = 0; mt < 2; ++mt) {
                const float* p = &As[(waveM * 32 + mt * 16 + l15) * LDA_S + ka];
                afr[mt].x = p[0];
                afr[mt].y = p[1];
            }
            #pragma unroll
            for (int nt = 0; nt < 2; ++nt) {
                const int col = waveN * 32 + nt * 16 + l15;
                bfr[nt].x = Bs[ka * BN + col];
                bfr[nt].y = Bs[(ka + 1) * BN + col];
            }
            #pragma unroll
            for (int mt = 0; mt < 2; ++mt)
                #pragma unroll
                for (int nt = 0; nt < 2; ++nt)
                    acc[mt][nt] = __builtin_amdgcn_wmma_f32_16x16x4_f32(
                        false, afr[mt], false, bfr[nt],
                        (short)0, acc[mt][nt], false, false);
        }
        __syncthreads();
    }

    float scale = 1.0f, bias = 0.0f;
    if (mode == 1) { scale = sc[0]; bias = bi[0]; }

    #pragma unroll
    for (int mt = 0; mt < 2; ++mt)
        #pragma unroll
        for (int nt = 0; nt < 2; ++nt) {
            const int n = bN + waveN * 32 + nt * 16 + l15;
            #pragma unroll
            for (int r = 0; r < 8; ++r) {
                const int m = bM + waveM * 32 + mt * 16 + r + half * 8;
                C[(size_t)m * Nc + n] = scale * acc[mt][nt][r] + bias;
            }
        }
}

// ===========================================================================
// vs -> (hi, lo) f16 split, 4 floats per thread
// ===========================================================================
__global__ __launch_bounds__(256)
void split_kernel(const float* __restrict__ src, _Float16* __restrict__ hi,
                  _Float16* __restrict__ lo)
{
    const size_t idx = (size_t)blockIdx.x * 256 + threadIdx.x;
    const float4 v = reinterpret_cast<const float4*>(src)[idx];
    h4 hh, hl;
    hh[0] = (_Float16)v.x; hl[0] = (_Float16)(v.x - (float)hh[0]);
    hh[1] = (_Float16)v.y; hl[1] = (_Float16)(v.y - (float)hh[1]);
    hh[2] = (_Float16)v.z; hl[2] = (_Float16)(v.z - (float)hh[2]);
    hh[3] = (_Float16)v.w; hl[3] = (_Float16)(v.w - (float)hh[3]);
    reinterpret_cast<h4*>(hi)[idx] = hh;
    reinterpret_cast<h4*>(lo)[idx] = hl;
}

// ===========================================================================
// S_t[c,r] = sigmoid(e1[r]*e2[c] + bs[r,c]) split into f16 hi/lo, TRANSPOSED
// output (shape [N,N], row = c, K-contiguous in r). 32x32 LDS tile transpose.
// ===========================================================================
__global__ __launch_bounds__(256)
void s_t_kernel(const float* __restrict__ e1, const float* __restrict__ e2,
                const float* __restrict__ bs,
                _Float16* __restrict__ st_hi, _Float16* __restrict__ st_lo)
{
    __shared__ _Float16 hti[32][33];
    __shared__ _Float16 tlo[32][33];
    const int tx = threadIdx.x & 31;
    const int ty = threadIdx.x >> 5;        // 0..7
    const int r0 = blockIdx.y * 32;
    const int c0 = blockIdx.x * 32;

    #pragma unroll
    for (int i = 0; i < 4; ++i) {
        const int r = r0 + ty + i * 8;
        const int c = c0 + tx;
        const float z = e1[r] * e2[c] + bs[(size_t)r * N_ + c];
        const float s = 1.0f / (1.0f + __expf(-z));
        const _Float16 h = (_Float16)s;
        hti[tx][ty + i * 8] = h;
        tlo[tx][ty + i * 8] = (_Float16)(s - (float)h);
    }
    __syncthreads();
    #pragma unroll
    for (int i = 0; i < 4; ++i) {
        const int c = c0 + ty + i * 8;      // output row
        const int r = r0 + tx;              // output col (K dim)
        st_hi[(size_t)c * N_ + r] = hti[ty + i * 8][tx];
        st_lo[(size_t)c * N_ + r] = tlo[ty + i * 8][tx];
    }
}

// ===========================================================================
// adj[m,:] = softmax_n(relu(emb[m].emb[n])) ; one block per row m
// ===========================================================================
__global__ __launch_bounds__(256)
void adj_kernel(const float* __restrict__ emb, float* __restrict__ adj)
{
    __shared__ float em[E_];
    __shared__ float zbuf[N_];
    __shared__ float red[256];
    const int m = blockIdx.x;
    const int tid = threadIdx.x;
    if (tid < E_) em[tid] = emb[m * E_ + tid];
    __syncthreads();

    float lmax = -1e30f;
    for (int n = tid; n < N_; n += 256) {
        float dot = 0.0f;
        #pragma unroll
        for (int e = 0; e < E_; ++e) dot += em[e] * emb[n * E_ + e];
        float z = fmaxf(dot, 0.0f);
        zbuf[n] = z;
        lmax = fmaxf(lmax, z);
    }
    red[tid] = lmax; __syncthreads();
    for (int s = 128; s > 0; s >>= 1) {
        if (tid < s) red[tid] = fmaxf(red[tid], red[tid + s]);
        __syncthreads();
    }
    const float rmax = red[0];
    __syncthreads();

    float lsum = 0.0f;
    for (int n = tid; n < N_; n += 256) {
        float ez = __expf(zbuf[n] - rmax);
        zbuf[n] = ez;
        lsum += ez;
    }
    red[tid] = lsum; __syncthreads();
    for (int s = 128; s > 0; s >>= 1) {
        if (tid < s) red[tid] += red[tid + s];
        __syncthreads();
    }
    const float inv = 1.0f / red[0];
    __syncthreads();
    for (int n = tid; n < N_; n += 256)
        adj[(size_t)m * N_ + n] = zbuf[n] * inv;
}

// ===========================================================================
__global__ __launch_bounds__(256)
void e1e2_kernel(const float* __restrict__ x, const float* __restrict__ w1,
                 const float* __restrict__ w2, float* __restrict__ e1,
                 float* __restrict__ e2)
{
    const int lane = threadIdx.x & 31;
    const int wave = threadIdx.x >> 5;
    const int row  = blockIdx.x * 8 + wave;
    const float* xr = x + (size_t)row * F_;
    const float xv0 = xr[lane];
    const float xv1 = xr[lane + 32];
    float a1 = xv0 * w1[lane] + xv1 * w1[lane + 32];
    float a2 = xv0 * w2[lane] + xv1 * w2[lane + 32];
    #pragma unroll
    for (int o = 16; o > 0; o >>= 1) {
        a1 += __shfl_xor(a1, o, 32);
        a2 += __shfl_xor(a2, o, 32);
    }
    if (lane == 0) { e1[row] = a1; e2[row] = a2; }
}

// ===========================================================================
__global__ __launch_bounds__(256)
void wmat_kernel(const float* __restrict__ w, const float* __restrict__ d,
                 float* __restrict__ W)
{
    __shared__ float dc[F_];
    const int tid = threadIdx.x;
    if (tid < F_) dc[tid] = fminf(fmaxf(d[tid], 0.0f), 1.0f);
    __syncthreads();
    for (int idx = tid; idx < F_ * F_; idx += 256) {
        const int i = idx >> 6, j = idx & 63;
        float s = 0.0f;
        for (int k = 0; k < F_; ++k) s += w[i * F_ + k] * dc[k] * w[j * F_ + k];
        W[idx] = s;
    }
}

// ===========================================================================
// in-place row softmax on T; one block per row
// ===========================================================================
__global__ __launch_bounds__(256)
void softmax_rows(float* __restrict__ T)
{
    __shared__ float red[256];
    const int m = blockIdx.x;
    const int tid = threadIdx.x;
    float* row = T + (size_t)m * N_;

    float v[16];
    float lmax = -1e30f;
    #pragma unroll
    for (int i = 0; i < 16; ++i) {
        v[i] = row[tid + i * 256];
        lmax = fmaxf(lmax, v[i]);
    }
    red[tid] = lmax; __syncthreads();
    for (int s = 128; s > 0; s >>= 1) {
        if (tid < s) red[tid] = fmaxf(red[tid], red[tid + s]);
        __syncthreads();
    }
    const float rmax = red[0]; __syncthreads();

    float lsum = 0.0f;
    #pragma unroll
    for (int i = 0; i < 16; ++i) {
        v[i] = __expf(v[i] - rmax);
        lsum += v[i];
    }
    red[tid] = lsum; __syncthreads();
    for (int s = 128; s > 0; s >>= 1) {
        if (tid < s) red[tid] += red[tid + s];
        __syncthreads();
    }
    const float inv = 1.0f / red[0];
    #pragma unroll
    for (int i = 0; i < 16; ++i) row[tid + i * 256] = v[i] * inv;
}

// ===========================================================================
// out = tanh(xe + 0.5*sigmoid(alpha)*xa + x@W + x0*sigmoid(beta) - 3x)
// ===========================================================================
__global__ __launch_bounds__(64)
void final_kernel(const float* __restrict__ x, const float* __restrict__ x0,
                  const float* __restrict__ alpha, const float* __restrict__ beta,
                  const float* __restrict__ Wm, const float* __restrict__ xa,
                  const float* __restrict__ xe, float* __restrict__ out)
{
    __shared__ float xs[F_];
    const int row = blockIdx.x;
    const int n   = row & (N_ - 1);
    const int f   = threadIdx.x;
    const size_t off = (size_t)row * F_ + f;
    const float xv = x[off];
    xs[f] = xv;
    __syncthreads();

    float acc = 0.0f;
    #pragma unroll 8
    for (int j = 0; j < F_; ++j) acc += xs[j] * Wm[j * F_ + f];

    const float sa = 1.0f / (1.0f + __expf(-alpha[n]));
    const float sb = 1.0f / (1.0f + __expf(-beta[n]));
    const float val = xe[off] + 0.5f * sa * xa[off] + acc + x0[off] * sb - 3.0f * xv;
    out[off] = tanhf(val);
}

// ===========================================================================
extern "C" void kernel_launch(void* const* d_in, const int* in_sizes, int n_in,
                              void* d_out, int out_size, void* d_ws, size_t ws_size,
                              hipStream_t stream)
{
    (void)in_sizes; (void)n_in; (void)out_size; (void)ws_size;

    const float* x     = (const float*)d_in[0];
    const float* x0    = (const float*)d_in[1];
    const float* alpha = (const float*)d_in[2];
    const float* beta  = (const float*)d_in[3];
    const float* w     = (const float*)d_in[4];
    const float* d     = (const float*)d_in[5];
    const float* w1    = (const float*)d_in[6];
    const float* w2    = (const float*)d_in[7];
    const float* vs    = (const float*)d_in[8];
    const float* bs    = (const float*)d_in[9];
    const float* emb   = (const float*)d_in[10];
    const float* convw = (const float*)d_in[11];
    const float* convb = (const float*)d_in[12];

    const size_t NN  = (size_t)N_ * N_;
    const size_t BNF = (size_t)B_ * N_ * F_;

    float*    ws_T  = (float*)d_ws;                 // adj first, then T (NN f32)
    _Float16* vs_hi = (_Float16*)(ws_T + NN);       // NN f16 each
    _Float16* vs_lo = vs_hi + NN;
    _Float16* st_hi = vs_lo + NN;
    _Float16* st_lo = st_hi + NN;
    float*    ws_xa = (float*)(st_lo + NN);
    float*    ws_xe = ws_xa + BNF;
    float*    ws_e1 = ws_xe + BNF;
    float*    ws_e2 = ws_e1 + (size_t)B_ * N_;
    float*    ws_W  = ws_e2 + (size_t)B_ * N_;

    wmat_kernel<<<1, 256, 0, stream>>>(w, d, ws_W);
    e1e2_kernel<<<(B_ * N_) / 8, 256, 0, stream>>>(x, w1, w2, ws_e1, ws_e2);
    adj_kernel<<<N_, 256, 0, stream>>>(emb, ws_T);
    split_kernel<<<(unsigned)(NN / 4 / 256), 256, 0, stream>>>(vs, vs_hi, vs_lo);

    const dim3 gsmall(F_ / BN, N_ / BM);   // (1, 32)  for Nc=64 f32 GEMMs
    const dim3 gbig(N_ / GN, N_ / GM);     // (32, 32) for f16x3 GEMM
    const dim3 gst(N_ / 32, N_ / 32);      // (128,128) sigmoid-transpose

    // xa[b] = conv_w * (adj @ x[b]) + conv_b   (adj currently in ws_T)
    for (int b = 0; b < B_; ++b)
        wmma_gemm_f32<<<gsmall, 256, 0, stream>>>(
            ws_T, x + (size_t)b * N_ * F_, ws_xa + (size_t)b * N_ * F_,
            N_, F_, N_, 1, convw, convb);

    // per batch: S_t(hi/lo) -> T = vs @ S (f16x3) -> softmax -> xe[b] = T @ x[b]
    for (int b = 0; b < B_; ++b) {
        s_t_kernel<<<gst, 256, 0, stream>>>(
            ws_e1 + (size_t)b * N_, ws_e2 + (size_t)b * N_, bs, st_hi, st_lo);
        wmma_gemm_f16x3<<<gbig, 256, 0, stream>>>(
            vs_hi, vs_lo, st_hi, st_lo, ws_T, N_, N_, N_);
        softmax_rows<<<N_, 256, 0, stream>>>(ws_T);
        wmma_gemm_f32<<<gsmall, 256, 0, stream>>>(
            ws_T, x + (size_t)b * N_ * F_, ws_xe + (size_t)b * N_ * F_,
            N_, F_, N_, 0, nullptr, nullptr);
    }

    final_kernel<<<B_ * N_, F_, 0, stream>>>(x, x0, alpha, beta, ws_W,
                                             ws_xa, ws_xe, (float*)d_out);
}
